// SingleHeadAttention_12481174962388
// MI455X (gfx1250) — matrix operand order repaired and checked by reading, not compiled, not run on hardware
//
#include <hip/hip_runtime.h>
#include <math.h>

typedef __attribute__((ext_vector_type(16))) _Float16 v16h;
typedef __attribute__((ext_vector_type(8)))  _Float16 v8h;
typedef __attribute__((ext_vector_type(8)))  float    v8f;
typedef __attribute__((ext_vector_type(4)))  float    f32x4;
typedef __attribute__((ext_vector_type(4)))  _Float16 v4h;

#define B_ 8
#define T_ 2048
#define E_ 1024
#define H_ 64

static __device__ inline v8f wmma_f16(v16h a, v16h b, v8f c) {
  return __builtin_amdgcn_wmma_f32_16x16x32_f16(
      /*neg_a=*/false, a, /*neg_b=*/false, b,
      /*c_mod=*/(short)0, c, /*reuse_a=*/false, /*reuse_b=*/false);
}

// A operand (16x32 f16): this lane's row, two contiguous 8-elem chunks (f32 src)
static __device__ inline v16h load_A_f32(const float* p0, const float* p1) {
  const f32x4* q0 = (const f32x4*)p0;
  const f32x4* q1 = (const f32x4*)p1;
  f32x4 a = q0[0], b = q0[1], c = q1[0], d = q1[1];
  v16h r;
#pragma unroll
  for (int i = 0; i < 4; i++) {
    r[i]      = (_Float16)a[i];
    r[4 + i]  = (_Float16)b[i];
    r[8 + i]  = (_Float16)c[i];
    r[12 + i] = (_Float16)d[i];
  }
  return r;
}

// A operand from f16 source: two contiguous 8-elem (16B) chunks
static __device__ inline v16h load_A_f16(const _Float16* p0, const _Float16* p1) {
  v8h a = *(const v8h*)p0;
  v8h b = *(const v8h*)p1;
  v16h r;
#pragma unroll
  for (int i = 0; i < 8; i++) { r[i] = a[i]; r[8 + i] = b[i]; }
  return r;
}

// ---------------- One-shot W fp32 -> f16 conversion ----------------
// Wh holds [Wq | Wk | Wv] as f16, each (H_, E_) row-major.
__global__ void __launch_bounds__(256)
sha_wcvt_kernel(const float* __restrict__ Wq,
                const float* __restrict__ Wk,
                const float* __restrict__ Wv,
                _Float16* __restrict__ Wh) {
  const int i = (blockIdx.x * 256 + threadIdx.x) * 4;  // 4 elems/thread
  const int n = H_ * E_;
  if (i >= n) return;
  f32x4 q = *(const f32x4*)(Wq + i);
  f32x4 k = *(const f32x4*)(Wk + i);
  f32x4 v = *(const f32x4*)(Wv + i);
  v4h hq, hk, hv;
#pragma unroll
  for (int j = 0; j < 4; j++) {
    hq[j] = (_Float16)q[j];
    hk[j] = (_Float16)k[j];
    hv[j] = (_Float16)v[j];
  }
  *(v4h*)(Wh + i)         = hq;
  *(v4h*)(Wh + n + i)     = hk;
  *(v4h*)(Wh + 2 * n + i) = hv;
}

// ---------------- Projection: Q = xWq^T, K = xWk^T, V^T ----------------
__global__ void __launch_bounds__(128)
sha_proj_kernel(const float* __restrict__ x,
                const _Float16* __restrict__ Wh,   // [Wq|Wk|Wv] f16
                _Float16* __restrict__ Qh,
                _Float16* __restrict__ Kh,
                _Float16* __restrict__ Vt) {
  const int lane = threadIdx.x & 31;
  const int wid  = threadIdx.x >> 5;
  const int hf   = lane >> 4;   // lane half: 0 or 1
  const int lr   = lane & 15;

  const int gw = blockIdx.x * 4 + wid;            // global wave id
  const int tilesPerB = T_ / 16;
  const int b  = gw / tilesPerB;
  const int t0 = (gw % tilesPerB) * 16;

  const _Float16* Wqh = Wh;
  const _Float16* Wkh = Wh + H_ * E_;
  const _Float16* Wvh = Wh + 2 * H_ * E_;

  v8f accQ[4], accK[4], accV[4];
#pragma unroll
  for (int i = 0; i < 4; i++) {
    accQ[i] = (v8f)0.0f; accK[i] = (v8f)0.0f; accV[i] = (v8f)0.0f;
  }

  const float* xrow = x + (size_t)(b * T_ + t0 + lr) * E_;

  for (int e0 = 0; e0 < E_; e0 += 32) {
    v16h a = load_A_f32(xrow + e0 + hf * 16, xrow + e0 + hf * 16 + 8);
#pragma unroll
    for (int ht = 0; ht < 4; ht++) {
      // B operand (32x16): column = W row (ht*16+lr), contiguous 16 f16 chunk
      const size_t woff = (size_t)(ht * 16 + lr) * E_ + e0 + hf * 16;
      accQ[ht] = wmma_f16(a, *(const v16h*)(Wqh + woff), accQ[ht]);
      accK[ht] = wmma_f16(a, *(const v16h*)(Wkh + woff), accK[ht]);
      accV[ht] = wmma_f16(a, *(const v16h*)(Wvh + woff), accV[ht]);
    }
  }

#pragma unroll
  for (int ht = 0; ht < 4; ht++) {
#pragma unroll
    for (int r = 0; r < 8; r++) {
      const int t = t0 + r + 8 * hf;     // C-layout: M = r + 8*half
      const int h = ht * 16 + lr;        // N = lane&15
      Qh[(size_t)(b * T_ + t) * H_ + h] = (_Float16)accQ[ht][r];
      Kh[(size_t)(b * T_ + t) * H_ + h] = (_Float16)accK[ht][r];
      Vt[(size_t)(b * H_ + h) * T_ + t] = (_Float16)accV[ht][r];  // transposed
    }
  }
}

// ---------------- Fused flash attention ----------------
__global__ void __launch_bounds__(128)
sha_attn_kernel(const _Float16* __restrict__ Qh,
                const _Float16* __restrict__ Kh,
                const _Float16* __restrict__ Vt,
                float* __restrict__ out,
                const int* __restrict__ should_mask_p) {
  __shared__ __align__(32) _Float16 pstage[4][16 * 32];  // per-wave P staging

  const int lane = threadIdx.x & 31;
  const int wid  = threadIdx.x >> 5;
  const int hf   = lane >> 4;
  const int lr   = lane & 15;

  const int gw = blockIdx.x * 4 + wid;
  const int tilesPerB = T_ / 16;
  const int b  = gw / tilesPerB;
  const int t0 = (gw % tilesPerB) * 16;

  const int mask = *should_mask_p;

  v8f o[4];
#pragma unroll
  for (int i = 0; i < 4; i++) o[i] = (v8f)0.0f;
  float mrow[8], lrow[8];
#pragma unroll
  for (int r = 0; r < 8; r++) { mrow[r] = -INFINITY; lrow[r] = 0.0f; }

  // Q tile as two A operands (H=64 -> k=32 x2), loaded once
  const _Float16* qrow = Qh + (size_t)(b * T_ + t0 + lr) * H_;
  const v16h aq0 = load_A_f16(qrow + hf * 16,      qrow + 8 + hf * 16);
  const v16h aq1 = load_A_f16(qrow + 32 + hf * 16, qrow + 40 + hf * 16);

  const int sEnd = mask ? (t0 + 16) : T_;

  for (int s0 = 0; s0 < sEnd; s0 += 32) {
    // S = Q K^T for two 16-wide key tiles
    v8f sA, sB;
    {
      const _Float16* k0 = Kh + (size_t)(b * T_ + s0 + lr) * H_ + hf * 16;
      v8f sacc = (v8f)0.0f;
      sacc = wmma_f16(aq0, *(const v16h*)k0, sacc);
      sacc = wmma_f16(aq1, *(const v16h*)(k0 + 32), sacc);
      sA = sacc;
    }
    {
      const _Float16* k1 = Kh + (size_t)(b * T_ + s0 + 16 + lr) * H_ + hf * 16;
      v8f sacc = (v8f)0.0f;
      sacc = wmma_f16(aq0, *(const v16h*)k1, sacc);
      sacc = wmma_f16(aq1, *(const v16h*)(k1 + 32), sacc);
      sB = sacc;
    }

    // scale by sqrt(H)=8 and apply causal mask in C-layout
#pragma unroll
    for (int r = 0; r < 8; r++) {
      sA[r] *= 8.0f;
      sB[r] *= 8.0f;
      if (mask) {
        const int t = t0 + r + 8 * hf;
        if (s0 + lr > t)      sA[r] = -INFINITY;
        if (s0 + 16 + lr > t) sB[r] = -INFINITY;
      }
    }

    // online softmax per row (row = lanes 0-15 of each half)
    _Float16* pw = pstage[wid];
#pragma unroll
    for (int r = 0; r < 8; r++) {
      float mx = fmaxf(sA[r], sB[r]);
#pragma unroll
      for (int off = 1; off < 16; off <<= 1)
        mx = fmaxf(mx, __shfl_xor(mx, off, 32));
      const float mnew  = fmaxf(mrow[r], mx);
      const float alpha = __expf(mrow[r] - mnew);
      mrow[r] = mnew;
      const float pa = __expf(sA[r] - mnew);
      const float pb = __expf(sB[r] - mnew);
      float rs = pa + pb;
#pragma unroll
      for (int off = 1; off < 16; off <<= 1)
        rs += __shfl_xor(rs, off, 32);
      lrow[r] = lrow[r] * alpha + rs;
#pragma unroll
      for (int ht = 0; ht < 4; ht++) o[ht][r] *= alpha;
      // stage P (C-layout -> row-major 16x32 in LDS)
      const int row = r + 8 * hf;
      pw[row * 32 + lr]      = (_Float16)pa;
      pw[row * 32 + 16 + lr] = (_Float16)pb;
    }

    // reload P in A layout (LDS ops are in-order within a wave)
    const _Float16* pr = pw + lr * 32;
  const v16h ap = load_A_f16(pr + hf * 16, pr + 8 + hf * 16);

    // O += P * V  (V^T rows give contiguous k=s chunks)
#pragma unroll
    for (int ht = 0; ht < 4; ht++) {
      const _Float16* vp =
          Vt + (size_t)(b * H_ + ht * 16 + lr) * T_ + s0 + hf * 16;
      o[ht] = wmma_f16(ap, *(const v16h*)vp, o[ht]);
    }
  }

  // finalize: divide by row sums, store fp32
#pragma unroll
  for (int r = 0; r < 8; r++) {
    const float inv = 1.0f / lrow[r];
    const int t = t0 + r + 8 * hf;
#pragma unroll
    for (int ht = 0; ht < 4; ht++) {
      out[(size_t)(b * T_ + t) * H_ + ht * 16 + lr] = o[ht][r] * inv;
    }
  }
}

extern "C" void kernel_launch(void* const* d_in, const int* in_sizes, int n_in,
                              void* d_out, int out_size, void* d_ws, size_t ws_size,
                              hipStream_t stream) {
  const float* x  = (const float*)d_in[0];
  const float* Wq = (const float*)d_in[1];
  const float* Wk = (const float*)d_in[2];
  const float* Wv = (const float*)d_in[3];
  const int* should_mask = (const int*)d_in[4];
  float* out = (float*)d_out;

  const size_t elems = (size_t)B_ * T_ * H_;
  _Float16* Qh = (_Float16*)d_ws;
  _Float16* Kh = Qh + elems;
  _Float16* Vt = Kh + elems;
  _Float16* Wh = Vt + elems;                // 3 * H_ * E_ f16 = 384 KB

  // 1) W -> f16 (one-shot, tiny)
  {
    const int n = H_ * E_;                  // per-matrix elements
    const int threads = n / 4;              // 4 elems/thread
    sha_wcvt_kernel<<<dim3((threads + 255) / 256), dim3(256), 0, stream>>>(
        Wq, Wk, Wv, Wh);
  }

  const int totalWaves = B_ * (T_ / 16);    // 1024
  const dim3 blk(128);                      // 4 waves/block
  const dim3 grid(totalWaves / 4);          // 256 blocks

  // 2) QKV projection (WMMA, f16 operands, fp32 accum)
  sha_proj_kernel<<<grid, blk, 0, stream>>>(x, Wh, Qh, Kh, Vt);
  // 3) fused flash attention (WMMA)
  sha_attn_kernel<<<grid, blk, 0, stream>>>(Qh, Kh, Vt, out, should_mask);
}
